// MultiHeadAttention_14508399526368
// MI455X (gfx1250) — matrix operand order repaired and checked by reading, not compiled, not run on hardware
//
#include <hip/hip_runtime.h>

// ---------------- Problem constants (match reference) ----------------
// B=4, S=2048, D=512, H=8, DK=DV=64
#define BB   4
#define SS   2048
#define DD   512
#define HH   8
#define DKV  64
#define LN_EPS 1e-5f

// ---------------- WMMA / TDM types (gfx1250, wave32) ----------------
typedef __bf16 bf16;
typedef bf16          v16bf __attribute__((ext_vector_type(16)));
typedef float         v8f   __attribute__((ext_vector_type(8)));
typedef unsigned int  v4u   __attribute__((ext_vector_type(4)));
typedef unsigned int  u32x4 __attribute__((ext_vector_type(4)));
typedef int           i32x4 __attribute__((ext_vector_type(4)));
typedef int           i32x8 __attribute__((ext_vector_type(8)));

union FragBF  { v16bf v; v4u u[2]; };
union PackBF4 { bf16 h[4]; uint2 u; };

// D = A(16x32 bf16) * B(32x16 bf16) + C(16x16 f32)
static __device__ __forceinline__ v8f wmma_bf16(const v16bf& a, const v16bf& b, const v8f& c) {
  return __builtin_amdgcn_wmma_f32_16x16x32_bf16(false, a, false, b, (short)0, c, false, false);
}

// ---------------------------------------------------------------------
// TDM: generic 2D tile load Global -> LDS (Tensor Data Mover).
//   dsz: 0=1B 1=2B elements. elems0 = contiguous tile length (elements),
//   rows = tile rows, srcStride = tensor row stride (elements),
//   tensorRows = tensor dim1 (for OOB). Optional LDS padding:
//   padInt code (interval = 2^(code+1) dwords), padAmt code (amount = code+1 dwords).
// Descriptor per CDNA5 ISA §8.3 (group0) / §8.4 (group1).
// ---------------------------------------------------------------------
static __device__ __forceinline__ void tdm_load_2d(
    unsigned ldsOff, const void* gptr, unsigned dsz,
    unsigned elems0, unsigned rows, unsigned srcStride, unsigned tensorRows,
    unsigned padEn, unsigned padInt, unsigned padAmt)
{
  unsigned long long ga = (unsigned long long)(size_t)gptr;
  u32x4 g0;
  g0[0] = 1u;                                   // count=1, user mode, no gather
  g0[1] = ldsOff;                               // lds_addr
  g0[2] = (unsigned)(ga & 0xffffffffu);         // global_addr lo
  g0[3] = (unsigned)((ga >> 32) & 0x1ffffffu)   // global_addr [56:32]
          | (2u << 30);                         // type = 2 ("image")
  i32x8 g1;
  g1[0] = (int)((dsz << 16) | (padEn << 20) | (padInt << 22) | (padAmt << 25));
  g1[1] = (int)((elems0 & 0xffffu) << 16);                          // tensor_dim0 lo16
  g1[2] = (int)(((elems0 >> 16) & 0xffffu) | ((tensorRows & 0xffffu) << 16));
  g1[3] = (int)(((tensorRows >> 16) & 0xffffu) | ((elems0 & 0xffffu) << 16)); // tile_dim0
  g1[4] = (int)(rows & 0xffffu);                                    // tile_dim1 (tile_dim2=0)
  g1[5] = (int)srcStride;                                           // tensor_dim0_stride lo32
  g1[6] = 0;
  g1[7] = 0;
  i32x4 gz = {0, 0, 0, 0};
#if defined(__clang_major__) && (__clang_major__ >= 23)
  i32x8 gz8 = {0, 0, 0, 0, 0, 0, 0, 0};
  __builtin_amdgcn_tensor_load_to_lds(g0, g1, gz, gz, gz8, 0);
#else
  __builtin_amdgcn_tensor_load_to_lds(g0, g1, gz, gz, 0);
#endif
}

// =====================================================================
// Kernel 1: QKV projection.  X[8192x512] @ W[512x512] -> bf16
//   grid.z: 0 -> Q ([B,H,S,DK]), 1 -> K (same), 2 -> V transposed [B,H,DV,S]
//   Tile 128(M) x 64(N); 8 waves, each 32x32. Double-buffered LDS staging.
// =====================================================================
__global__ __launch_bounds__(256)
void k_qkv_proj(const float* __restrict__ Xq, const float* __restrict__ Xk,
                const float* __restrict__ Xv, const float* __restrict__ Wq,
                const float* __restrict__ Wk, const float* __restrict__ Wv,
                bf16* __restrict__ Qo, bf16* __restrict__ Ko, bf16* __restrict__ Vt)
{
  const int mode = blockIdx.z;
  const float* X = (mode == 0) ? Xq : (mode == 1) ? Xk : Xv;
  const float* W = (mode == 0) ? Wq : (mode == 1) ? Wk : Wv;
  bf16* Out      = (mode == 0) ? Qo : (mode == 1) ? Ko : Vt;

  __shared__ bf16 As[2][128][40];   // ping-pong, 80B row stride
  __shared__ bf16 Bt[2][64][48];    // W tile transposed

  const int t    = threadIdx.x;
  const int lane = t & 31, wv = t >> 5;
  const int l16  = lane & 15, lh = lane >> 4;
  const int wm   = wv >> 1, wn = wv & 1;
  const int rowBase = blockIdx.x * 128;
  const int colBase = blockIdx.y * 64;

  auto stage = [&](int buf, int k0) {
    // A: 128x32 fp32 -> bf16 (two-phase: clause loads, then convert+store)
    float4 fa[4];
    int ra[4], ca[4];
    for (int i = 0; i < 4; ++i) {
      int e = i * 256 + t; ra[i] = e >> 3; ca[i] = (e & 7) * 4;
      fa[i] = *(const float4*)&X[(size_t)(rowBase + ra[i]) * DD + k0 + ca[i]];
    }
    // W: 32x64 fp32 -> bf16 transposed
    float4 fw[2];
    int kw[2], nw[2];
    for (int i = 0; i < 2; ++i) {
      int e = i * 256 + t; kw[i] = e >> 4; nw[i] = (e & 15) * 4;
      fw[i] = *(const float4*)&W[(size_t)(k0 + kw[i]) * DD + colBase + nw[i]];
    }
    for (int i = 0; i < 4; ++i) {
      PackBF4 pk; pk.h[0] = (bf16)fa[i].x; pk.h[1] = (bf16)fa[i].y;
                  pk.h[2] = (bf16)fa[i].z; pk.h[3] = (bf16)fa[i].w;
      *(uint2*)&As[buf][ra[i]][ca[i]] = pk.u;
    }
    for (int i = 0; i < 2; ++i) {
      Bt[buf][nw[i] + 0][kw[i]] = (bf16)fw[i].x;
      Bt[buf][nw[i] + 1][kw[i]] = (bf16)fw[i].y;
      Bt[buf][nw[i] + 2][kw[i]] = (bf16)fw[i].z;
      Bt[buf][nw[i] + 3][kw[i]] = (bf16)fw[i].w;
    }
  };

  v8f acc[2][2];
  for (int i = 0; i < 2; ++i)
    for (int j = 0; j < 2; ++j) acc[i][j] = v8f{0.f,0.f,0.f,0.f,0.f,0.f,0.f,0.f};

  stage(0, 0);
  __syncthreads();

  for (int kt = 0; kt < 16; ++kt) {
    const int cur = kt & 1, nxt = cur ^ 1;
    if (kt < 15) stage(nxt, (kt + 1) * 32);

    FragBF a[2], bfr[2];
    for (int i = 0; i < 2; ++i) {
      int r = wm * 32 + 16 * i + l16, kb = lh * 8;
      a[i].u[0] = *(const v4u*)&As[cur][r][kb];
      a[i].u[1] = *(const v4u*)&As[cur][r][kb + 16];
    }
    for (int j = 0; j < 2; ++j) {
      int n = wn * 32 + 16 * j + l16, kb = lh * 8;
      bfr[j].u[0] = *(const v4u*)&Bt[cur][n][kb];
      bfr[j].u[1] = *(const v4u*)&Bt[cur][n][kb + 16];
    }
    for (int i = 0; i < 2; ++i)
      for (int j = 0; j < 2; ++j)
        acc[i][j] = wmma_bf16(a[i].v, bfr[j].v, acc[i][j]);
    __syncthreads();
  }

  for (int i = 0; i < 2; ++i)
    for (int j = 0; j < 2; ++j)
      for (int r = 0; r < 8; ++r) {
        int gr = rowBase + wm * 32 + 16 * i + 8 * lh + r;
        int gc = colBase + wn * 32 + 16 * j + l16;
        int b = gr >> 11, s = gr & (SS - 1);
        int h = gc >> 6,  d = gc & (DKV - 1);
        float vv = acc[i][j][r];
        if (mode < 2)
          Out[(((size_t)(b * HH + h)) * SS + s) * DKV + d] = (bf16)vv;
        else
          Out[(((size_t)(b * HH + h)) * DKV + d) * SS + s] = (bf16)vv;   // V^T
      }
}

// =====================================================================
// Kernel 2: scores = Q @ K^T * 1/8, masked.
//   Q tile (128x64 bf16), K tile (128x64 bf16) and mask tile (128x128 u8)
//   all fetched by the Tensor Data Mover; inner loop runs from LDS only.
//   Q/K tiles padded 16B per 128B row -> 144B stride, conflict-free frags.
// =====================================================================
__global__ __launch_bounds__(256)
void k_scores(const bf16* __restrict__ Q, const bf16* __restrict__ K,
              const unsigned char* __restrict__ mask, float* __restrict__ attn)
{
  __shared__ bf16 Qs[128][72];            // 144B row stride
  __shared__ bf16 Ks[128][72];
  __shared__ unsigned char Msk[128][128]; // unpadded (byte reads)

  const int t = threadIdx.x, lane = t & 31, wv = t >> 5;
  const int l16 = lane & 15, lh = lane >> 4;
  const int wm = wv >> 1, wn = wv & 1;
  const int bh = blockIdx.z, b = bh >> 3;
  const int qb = blockIdx.x * 128;
  const int nb = blockIdx.y * 128;
  const bf16* Qh = Q + (size_t)bh * SS * DKV;
  const bf16* Kh = K + (size_t)bh * SS * DKV;

  if (t < 32) {
    // Q rows [qb, qb+128): tile 64 x 128, pad 4 dwords every 32 dwords (one row)
    tdm_load_2d((unsigned)(size_t)&Qs[0][0], Qh + (size_t)qb * DKV, /*dsz=*/1,
                DKV, 128, DKV, SS, /*pad*/1, /*int:32dw*/4, /*amt:4dw*/3);
    // K rows [nb, nb+128)
    tdm_load_2d((unsigned)(size_t)&Ks[0][0], Kh + (size_t)nb * DKV, /*dsz=*/1,
                DKV, 128, DKV, SS, 1, 4, 3);
    // mask tile 128x128 bytes (no padding)
    tdm_load_2d((unsigned)(size_t)&Msk[0][0],
                mask + ((size_t)b * SS + qb) * SS + nb, /*dsz=*/0,
                128, 128, SS, BB * SS, 0, 0, 0);
    __builtin_amdgcn_s_wait_tensorcnt(0);
  }
  __syncthreads();

  v8f acc[2][4];
  for (int i = 0; i < 2; ++i)
    for (int j = 0; j < 4; ++j) acc[i][j] = v8f{0.f,0.f,0.f,0.f,0.f,0.f,0.f,0.f};

  for (int ks = 0; ks < 2; ++ks) {
    const int kofs = ks * 32;
    FragBF a[2], bb[4];
    for (int i = 0; i < 2; ++i) {
      int q = wm * 32 + 16 * i + l16, kb = kofs + lh * 8;
      a[i].u[0] = *(const v4u*)&Qs[q][kb];
      a[i].u[1] = *(const v4u*)&Qs[q][kb + 16];
    }
    for (int j = 0; j < 4; ++j) {
      int n = wn * 64 + 16 * j + l16, kb = kofs + lh * 8;
      bb[j].u[0] = *(const v4u*)&Ks[n][kb];
      bb[j].u[1] = *(const v4u*)&Ks[n][kb + 16];
    }
    for (int i = 0; i < 2; ++i)
      for (int j = 0; j < 4; ++j)
        acc[i][j] = wmma_bf16(a[i].v, bb[j].v, acc[i][j]);
  }

  const float scale = 0.125f;  // 1/sqrt(64)
  for (int i = 0; i < 2; ++i)
    for (int j = 0; j < 4; ++j)
      for (int r = 0; r < 8; ++r) {
        int ql = wm * 32 + 16 * i + 8 * lh + r;
        int kl = wn * 64 + 16 * j + l16;
        float vv = acc[i][j][r] * scale;
        if (Msk[ql][kl]) vv = -1e9f;
        attn[((size_t)bh * SS + qb + ql) * SS + nb + kl] = vv;
      }
}

// =====================================================================
// Kernel 3: row softmax over 2048 elems (vectorized b128 row I/O).
// =====================================================================
__global__ __launch_bounds__(256)
void k_softmax(float* __restrict__ attn)
{
  __shared__ float red[8];
  const int t = threadIdx.x, lane = t & 31, wv = t >> 5;
  float* row = attn + (size_t)blockIdx.x * SS;

  float4 f0 = *(const float4*)&row[t * 8];
  float4 f1 = *(const float4*)&row[t * 8 + 4];
  float v[8] = {f0.x, f0.y, f0.z, f0.w, f1.x, f1.y, f1.z, f1.w};

  float m = -3.0e38f;
  for (int i = 0; i < 8; ++i) m = fmaxf(m, v[i]);
  for (int o = 16; o; o >>= 1) m = fmaxf(m, __shfl_xor(m, o, 32));
  if (lane == 0) red[wv] = m;
  __syncthreads();
  m = red[lane & 7];
  for (int o = 4; o; o >>= 1) m = fmaxf(m, __shfl_xor(m, o, 32));

  float s = 0.f;
  for (int i = 0; i < 8; ++i) { v[i] = __expf(v[i] - m); s += v[i]; }
  for (int o = 16; o; o >>= 1) s += __shfl_xor(s, o, 32);
  __syncthreads();
  if (lane == 0) red[wv] = s;
  __syncthreads();
  s = red[lane & 7];
  for (int o = 4; o; o >>= 1) s += __shfl_xor(s, o, 32);

  const float inv = 1.0f / s;
  float4 o0 = {v[0] * inv, v[1] * inv, v[2] * inv, v[3] * inv};
  float4 o1 = {v[4] * inv, v[5] * inv, v[6] * inv, v[7] * inv};
  *(float4*)&row[t * 8]     = o0;
  *(float4*)&row[t * 8 + 4] = o1;
}

// =====================================================================
// Kernel 4: context = attn @ V.  attn staged fp32->bf16, double-buffered,
//   with global_prefetch of the next K-tile; V read via V^T (contiguous).
// =====================================================================
__global__ __launch_bounds__(256)
void k_context(const float* __restrict__ attn, const bf16* __restrict__ Vt,
               bf16* __restrict__ Ctx)
{
  __shared__ bf16 As[2][128][40];
  const int t = threadIdx.x, lane = t & 31, wv = t >> 5;
  const int l16 = lane & 15, lh = lane >> 4;
  const int wm = wv >> 1, wn = wv & 1;
  const int bh = blockIdx.z, b = bh >> 3, h = bh & 7;
  const int q0 = blockIdx.x * 128;
  const float* Ah = attn + (size_t)bh * SS * SS;
  const bf16*  Vh = Vt   + (size_t)bh * DKV * SS;

  auto stage = [&](int buf, int k0) {
    float4 f[4];
    int ra[4], ca[4];
    for (int i = 0; i < 4; ++i) {
      int e = i * 256 + t; ra[i] = e >> 3; ca[i] = (e & 7) * 4;
      const float* src = &Ah[(size_t)(q0 + ra[i]) * SS + k0 + ca[i]];
      f[i] = *(const float4*)src;
      if (k0 + 32 < SS) __builtin_prefetch(src + 32, 0, 1);
    }
    for (int i = 0; i < 4; ++i) {
      PackBF4 pk; pk.h[0] = (bf16)f[i].x; pk.h[1] = (bf16)f[i].y;
                  pk.h[2] = (bf16)f[i].z; pk.h[3] = (bf16)f[i].w;
      *(uint2*)&As[buf][ra[i]][ca[i]] = pk.u;
    }
  };

  v8f acc[2][2];
  for (int i = 0; i < 2; ++i)
    for (int j = 0; j < 2; ++j) acc[i][j] = v8f{0.f,0.f,0.f,0.f,0.f,0.f,0.f,0.f};

  stage(0, 0);
  __syncthreads();

  for (int kt = 0; kt < 64; ++kt) {
    const int k0 = kt * 32;
    const int cur = kt & 1, nxt = cur ^ 1;
    if (kt < 63) stage(nxt, k0 + 32);

    FragBF a[2], bb[2];
    for (int i = 0; i < 2; ++i) {
      int r = wm * 32 + 16 * i + l16, kb = lh * 8;
      a[i].u[0] = *(const v4u*)&As[cur][r][kb];
      a[i].u[1] = *(const v4u*)&As[cur][r][kb + 16];
    }
    for (int j = 0; j < 2; ++j) {
      int n = wn * 32 + 16 * j + l16, kb = lh * 8;
      bb[j].u[0] = *(const v4u*)(Vh + (size_t)n * SS + k0 + kb);
      bb[j].u[1] = *(const v4u*)(Vh + (size_t)n * SS + k0 + kb + 16);
    }
    for (int i = 0; i < 2; ++i)
      for (int j = 0; j < 2; ++j)
        acc[i][j] = wmma_bf16(a[i].v, bb[j].v, acc[i][j]);
    __syncthreads();
  }

  for (int i = 0; i < 2; ++i)
    for (int j = 0; j < 2; ++j)
      for (int r = 0; r < 8; ++r) {
        int qg = q0 + wm * 32 + 16 * i + 8 * lh + r;
        int dv = wn * 32 + 16 * j + l16;
        Ctx[((size_t)b * SS + qg) * DD + h * DKV + dv] = (bf16)acc[i][j][r];
      }
}

// =====================================================================
// Kernel 5: out = LayerNorm(Ctx @ W_fc + input_Q).
//   A-slab (32x512 bf16) via one TDM DMA with 16B/row padding (stride 520);
//   W_fc K-slices staged transposed, double-buffered. Fused residual + LN.
// =====================================================================
__global__ __launch_bounds__(256)
void k_fc_ln(const bf16* __restrict__ Ctx, const float* __restrict__ Wfc,
             const float* __restrict__ Xq, const float* __restrict__ gamma,
             const float* __restrict__ beta, float* __restrict__ out)
{
  __shared__ bf16  As[32][520];      // filled by TDM (row stride 1040B)
  __shared__ bf16  Bt[2][512][40];   // W_fc K-slice transposed, ping-pong
  __shared__ float rsum[32][8];
  __shared__ float rsumsq[32][8];
  __shared__ float rmean[32], rrstd[32];

  const int t = threadIdx.x, lane = t & 31, wv = t >> 5;
  const int l16 = lane & 15, lh = lane >> 4;
  const int r0 = blockIdx.x * 32;
  const int nW = wv * 64;

  auto stageB = [&](int buf, int k0) {
    for (int g = 0; g < 4; ++g) {          // 4 chunks of 4 loads (VGPR pressure)
      float4 f[4];
      int kr[4], nc[4];
      for (int u = 0; u < 4; ++u) {
        int e = (g * 4 + u) * 256 + t; kr[u] = e >> 7; nc[u] = (e & 127) * 4;
        f[u] = *(const float4*)&Wfc[(size_t)(k0 + kr[u]) * DD + nc[u]];
      }
      for (int u = 0; u < 4; ++u) {
        Bt[buf][nc[u] + 0][kr[u]] = (bf16)f[u].x;
        Bt[buf][nc[u] + 1][kr[u]] = (bf16)f[u].y;
        Bt[buf][nc[u] + 2][kr[u]] = (bf16)f[u].z;
        Bt[buf][nc[u] + 3][kr[u]] = (bf16)f[u].w;
      }
    }
  };

  // One TDM descriptor: tile 512(K) x 32(rows) of bf16 from Ctx row r0.
  if (t < 32)
    tdm_load_2d((unsigned)(size_t)&As[0][0], Ctx + (size_t)r0 * DD, /*dsz=*/1,
                DD, 32, DD, BB * SS, /*pad*/1, /*int:256dw*/7, /*amt:4dw*/3);
  stageB(0, 0);
  if (t < 32) __builtin_amdgcn_s_wait_tensorcnt(0);
  __syncthreads();

  v8f acc[2][4];
  for (int i = 0; i < 2; ++i)
    for (int j = 0; j < 4; ++j) acc[i][j] = v8f{0.f,0.f,0.f,0.f,0.f,0.f,0.f,0.f};

  for (int kt = 0; kt < 16; ++kt) {
    const int k0 = kt * 32;
    const int cur = kt & 1, nxt = cur ^ 1;
    if (kt < 15) stageB(nxt, k0 + 32);

    FragBF a[2], bb[4];
    for (int i = 0; i < 2; ++i) {
      int r = 16 * i + l16, kb = k0 + lh * 8;
      a[i].u[0] = *(const v4u*)&As[r][kb];
      a[i].u[1] = *(const v4u*)&As[r][kb + 16];
    }
    for (int j = 0; j < 4; ++j) {
      int n = nW + 16 * j + l16, kb = lh * 8;
      bb[j].u[0] = *(const v4u*)&Bt[cur][n][kb];
      bb[j].u[1] = *(const v4u*)&Bt[cur][n][kb + 16];
    }
    for (int i = 0; i < 2; ++i)
      for (int j = 0; j < 4; ++j)
        acc[i][j] = wmma_bf16(a[i].v, bb[j].v, acc[i][j]);
    __syncthreads();
  }

  // residual + per-row partial stats (this wave's 64 columns)
  for (int i = 0; i < 2; ++i)
    for (int r = 0; r < 8; ++r) {
      const int row = 16 * i + 8 * lh + r;
      float s = 0.f, ssq = 0.f;
      for (int j = 0; j < 4; ++j) {
        int col = nW + 16 * j + l16;
        float val = acc[i][j][r] + Xq[(size_t)(r0 + row) * DD + col];
        acc[i][j][r] = val;
        s += val; ssq += val * val;
      }
      for (int o = 8; o; o >>= 1) { s += __shfl_xor(s, o, 32); ssq += __shfl_xor(ssq, o, 32); }
      if (l16 == 0) { rsum[row][wv] = s; rsumsq[row][wv] = ssq; }
    }
  __syncthreads();

  if (t < 32) {
    float s = 0.f, ssq = 0.f;
    for (int w = 0; w < 8; ++w) { s += rsum[t][w]; ssq += rsumsq[t][w]; }
    float mean = s * (1.f / (float)DD);
    float var  = ssq * (1.f / (float)DD) - mean * mean;
    rmean[t] = mean;
    rrstd[t] = rsqrtf(var + LN_EPS);
  }
  __syncthreads();

  for (int i = 0; i < 2; ++i)
    for (int r = 0; r < 8; ++r) {
      const int row = 16 * i + 8 * lh + r;
      const float mean = rmean[row], rstd = rrstd[row];
      for (int j = 0; j < 4; ++j) {
        int col = nW + 16 * j + l16;
        float val = (acc[i][j][r] - mean) * rstd * gamma[col] + beta[col];
        out[(size_t)(r0 + row) * DD + col] = val;
      }
    }
}

// =====================================================================
// Host-side launcher
// =====================================================================
extern "C" void kernel_launch(void* const* d_in, const int* in_sizes, int n_in,
                              void* d_out, int out_size, void* d_ws, size_t ws_size,
                              hipStream_t stream) {
  (void)in_sizes; (void)n_in; (void)out_size; (void)ws_size;

  const float* Xq = (const float*)d_in[0];
  const float* Xk = (const float*)d_in[1];
  const float* Xv = (const float*)d_in[2];
  const unsigned char* mask = (const unsigned char*)d_in[3];  // jnp bool -> 1 byte/elem
  const float* Wq  = (const float*)d_in[4];
  const float* Wk  = (const float*)d_in[5];
  const float* Wv  = (const float*)d_in[6];
  const float* Wfc = (const float*)d_in[7];
  const float* gam = (const float*)d_in[8];
  const float* bet = (const float*)d_in[9];

  float* out  = (float*)d_out;                         // normed: B*S*D
  float* attn = out + (size_t)BB * SS * DD;            // attn:   B*H*S*S

  const size_t headElems = (size_t)BB * HH * SS * DKV; // 4,194,304
  bf16* Qbf = (bf16*)d_ws;
  bf16* Kbf = Qbf + headElems;
  bf16* Vtb = Kbf + headElems;
  bf16* Ctx = Vtb + headElems;                         // total ws use: 32 MB

  dim3 blk(256);
  k_qkv_proj<<<dim3(SS * BB / 128, DD / 64, 3), blk, 0, stream>>>(
      Xq, Xk, Xv, Wq, Wk, Wv, Qbf, Kbf, Vtb);
  k_scores<<<dim3(SS / 128, SS / 128, BB * HH), blk, 0, stream>>>(
      Qbf, Kbf, mask, attn);
  k_softmax<<<dim3(BB * HH * SS), blk, 0, stream>>>(attn);
  k_context<<<dim3(SS / 128, 1, BB * HH), blk, 0, stream>>>(attn, Vtb, Ctx);
  k_fc_ln<<<dim3(BB * SS / 32), blk, 0, stream>>>(Ctx, Wfc, Xq, gam, bet, out);
}
